// RelativeMultiHeadAttention_42786464202902
// MI455X (gfx1250) — matrix operand order repaired and checked by reading, not compiled, not run on hardware
//
#include <hip/hip_runtime.h>
#include <hip/hip_bf16.h>
#include <math.h>

// ---------------------------------------------------------------------------
// CDNA5 / gfx1250 WMMA types (wave32). 16x16x32 bf16 -> f32 accumulate.
// ---------------------------------------------------------------------------
typedef __attribute__((ext_vector_type(16))) __bf16 v16bf;
typedef __attribute__((ext_vector_type(8)))  float  v8f;

union FragBF { v16bf v; __bf16 e[16]; };
union FragF  { v8f   v; float  e[8];  };

#define D_MODEL 512
#define N_HEADS 16
#define HEAD_DIM 32
#define BATCH 2
#define SEQ 2048
#define HALF (D_MODEL / 2)

// ---------------------------------------------------------------------------
// Async global->LDS copy (CDNA5 ASYNCcnt path), guarded so either toolchain
// compiles. The builtin takes (AS1 int4*, AS3 int4*, imm offset, imm cpol)
// per the hipcc diagnostic. Fallback: global_load_b128 + ds_store_b128.
// ---------------------------------------------------------------------------
#if __has_builtin(__builtin_amdgcn_global_load_async_to_lds_b128)
#define HAS_ASYNC_COPY 1
#else
#define HAS_ASYNC_COPY 0
#endif
#if __has_builtin(__builtin_amdgcn_s_wait_asynccnt)
#define HAS_ASYNC_WAIT 1
#else
#define HAS_ASYNC_WAIT 0
#endif

typedef int v4i_t __attribute__((vector_size(16)));

__device__ __forceinline__ void copy16_g2l(void* lds, const void* g) {
#if HAS_ASYNC_COPY
    __builtin_amdgcn_global_load_async_to_lds_b128(
        (__attribute__((address_space(1))) v4i_t*)g,
        (__attribute__((address_space(3))) v4i_t*)lds, 0, 0);
#else
    *(uint4*)lds = *(const uint4*)g;
#endif
}

__device__ __forceinline__ void async_fence_lds() {
#if HAS_ASYNC_COPY
#if HAS_ASYNC_WAIT
    __builtin_amdgcn_s_wait_asynccnt(0);
#else
    asm volatile("s_wait_asynccnt 0x0" ::: "memory");
#endif
#endif
}

// ---------------------------------------------------------------------------
// f32 -> bf16 flat conversion
// ---------------------------------------------------------------------------
__global__ void f32_to_bf16_kernel(const float* __restrict__ in,
                                   __bf16* __restrict__ out, int n) {
    int i = blockIdx.x * blockDim.x + threadIdx.x;
    if (i < n) out[i] = (__bf16)in[i];
}

// ---------------------------------------------------------------------------
// Fused RoPE + head split/pack.
//   qh,kh,vh : f32 [B,T,D]
//   Qb,Kb    : bf16 [B,H,T,HD]   (rotated)
//   Vt       : bf16 [B,H,HD,T]   (transposed so P*V B-fragments are contiguous)
// ---------------------------------------------------------------------------
__global__ void rope_split_kernel(const float* __restrict__ qh,
                                  const float* __restrict__ kh,
                                  const float* __restrict__ vh,
                                  const float* __restrict__ cosT,
                                  const float* __restrict__ sinT,
                                  __bf16* __restrict__ Qb,
                                  __bf16* __restrict__ Kb,
                                  __bf16* __restrict__ Vt) {
    int idx = blockIdx.x * blockDim.x + threadIdx.x;
    if (idx >= BATCH * SEQ * HALF) return;
    int i = idx % HALF;
    int t = (idx / HALF) % SEQ;
    int b = idx / (HALF * SEQ);
    int d0 = 2 * i;
    int h  = d0 / HEAD_DIM;
    int hd = d0 % HEAD_DIM;

    float c = cosT[t * HALF + i];
    float s = sinT[t * HALF + i];

    size_t src = ((size_t)(b * SEQ + t)) * D_MODEL + d0;
    float q0 = qh[src], q1 = qh[src + 1];
    float k0 = kh[src], k1 = kh[src + 1];
    float v0 = vh[src], v1 = vh[src + 1];

    size_t dstQ = (((size_t)(b * N_HEADS + h)) * SEQ + t) * HEAD_DIM + hd;
    Qb[dstQ]     = (__bf16)(q0 * c - q1 * s);
    Qb[dstQ + 1] = (__bf16)(q0 * s + q1 * c);
    Kb[dstQ]     = (__bf16)(k0 * c - k1 * s);
    Kb[dstQ + 1] = (__bf16)(k0 * s + k1 * c);

    size_t dstV = (((size_t)(b * N_HEADS + h)) * HEAD_DIM + hd) * SEQ + t;
    Vt[dstV]       = (__bf16)v0;
    Vt[dstV + SEQ] = (__bf16)v1;
}

// ---------------------------------------------------------------------------
// C[M,N] = A[M,K] * W[N,K]^T + bias[N]    (bf16 inputs, f32 output)
// Block: 256 threads = 8 waves (2x4). Wave tile: 32x64 = 2 A-frags x 4 B-frags
// -> 8 WMMAs per 32-K step, each B fragment reused twice. All fragment loads
// grouped ahead of the WMMA burst so loadcnt waits batch and the next
// iteration's loads overlap current WMMAs.
// Fragment packing uses one contiguous k order (k = kbase + s) for both A and B:
//   A 16x32: lane m=lane&15; K-base 0 (lanes 0-15) / 16 (lanes 16-31);
//            VGPR v holds K = kbase+2v, +1.
//   B 32x16: lane n=lane&15; K-base 0/16; VGPR v holds K = kbase+2v, +1.
// ---------------------------------------------------------------------------
__global__ void __launch_bounds__(256)
gemm_bf16_wmma_kernel(const __bf16* __restrict__ A,
                      const __bf16* __restrict__ W,
                      const float* __restrict__ bias,
                      float* __restrict__ C,
                      int M, int N, int K) {
    int lane = threadIdx.x & 31;
    int wave = threadIdx.x >> 5;
    int wy = wave >> 2;          // 0..1
    int wx = wave & 3;           // 0..3
    int m0 = blockIdx.y * 64 + wy * 32;
    int n0 = blockIdx.x * 256 + wx * 64;

    int lm = lane & 15;
    int ka = (lane < 16) ? 0 : 16;   // A-fragment K base
    int kb = (lane < 16) ? 0 : 16;   // B-fragment K base

    FragF acc[2][4];
#pragma unroll
    for (int i = 0; i < 2; ++i)
#pragma unroll
        for (int j = 0; j < 4; ++j)
#pragma unroll
            for (int r = 0; r < 8; ++r) acc[i][j].e[r] = 0.0f;

    for (int k0 = 0; k0 < K; k0 += 32) {
        FragBF a0, a1;
        const __bf16* Ar0 = A + (size_t)(m0 + lm) * K + k0;
        const __bf16* Ar1 = A + (size_t)(m0 + 16 + lm) * K + k0;
#pragma unroll
        for (int v2 = 0; v2 < 8; ++v2) {
            int kk = 2 * v2 + ka;
            a0.e[2 * v2] = Ar0[kk]; a0.e[2 * v2 + 1] = Ar0[kk + 1];
            a1.e[2 * v2] = Ar1[kk]; a1.e[2 * v2 + 1] = Ar1[kk + 1];
        }
        if (k0 + 32 < K) {
            __builtin_prefetch(Ar0 + 32, 0, 1);
            __builtin_prefetch(Ar1 + 32, 0, 1);
        }

        FragBF bfr[4];
#pragma unroll
        for (int j = 0; j < 4; ++j) {
            const __bf16* Wrow = W + (size_t)(n0 + j * 16 + lm) * K + k0 + kb;
#pragma unroll
            for (int v2 = 0; v2 < 8; ++v2) {
                bfr[j].e[2 * v2]     = Wrow[2 * v2];
                bfr[j].e[2 * v2 + 1] = Wrow[2 * v2 + 1];
            }
        }

#pragma unroll
        for (int j = 0; j < 4; ++j) {
            acc[0][j].v = __builtin_amdgcn_wmma_f32_16x16x32_bf16(
                false, a0.v, false, bfr[j].v, (short)0, acc[0][j].v, false, false);
            acc[1][j].v = __builtin_amdgcn_wmma_f32_16x16x32_bf16(
                false, a1.v, false, bfr[j].v, (short)0, acc[1][j].v, false, false);
        }
    }

    // C/D layout: lane n=lane&15; rows r (lanes 0-15) / 8+r (lanes 16-31)
    int mb = (lane < 16) ? 0 : 8;
#pragma unroll
    for (int i = 0; i < 2; ++i) {
        int mrow0 = m0 + i * 16 + mb;
#pragma unroll
        for (int j = 0; j < 4; ++j) {
            int n = n0 + j * 16 + lm;
            float bs = bias[n];
#pragma unroll
            for (int r = 0; r < 8; ++r)
                C[(size_t)(mrow0 + r) * N + n] = acc[i][j].e[r] + bs;
        }
    }
}

// ---------------------------------------------------------------------------
// Flash attention, one wave per 16-query tile of one (b,h).
//   Q,Kb : bf16 [B,H,T,HD],  Vt : bf16 [B,H,HD,T],  ctx : f32 [B,T,D]
// Per 32-key step: K/V tiles staged into LDS via async global->LDS copies
// (ASYNCcnt path when available), 2 WMMAs for S, online softmax (shfl_xor over
// 16-lane halves), P bounced through LDS (C-layout f32 -> A-layout bf16),
// 2 WMMAs for O += P*V. T and HD are multiples of the WMMA shapes -> no tails.
// ---------------------------------------------------------------------------
#define ATT_WAVES 4

__global__ void __launch_bounds__(32 * ATT_WAVES)
flash_attn_kernel(const __bf16* __restrict__ Q,
                  const __bf16* __restrict__ Kb,
                  const __bf16* __restrict__ Vt,
                  float* __restrict__ ctx) {
    __shared__ float  pbuf[ATT_WAVES][16][32];
    __shared__ __bf16 ktile[ATT_WAVES][32][32];   // [key][dim]
    __shared__ __bf16 vtile[ATT_WAVES][32][32];   // [dim][key]

    int lane = threadIdx.x & 31;
    int wave = threadIdx.x >> 5;
    int tile = blockIdx.x * ATT_WAVES + wave;
    int tilesPerBH = SEQ / 16;
    int bh = tile / tilesPerBH;
    int m0 = (tile % tilesPerBH) * 16;
    int b = bh / N_HEADS;
    int h = bh % N_HEADS;

    const __bf16* Qh = Q  + (size_t)bh * SEQ * HEAD_DIM;
    const __bf16* Kh = Kb + (size_t)bh * SEQ * HEAD_DIM;
    const __bf16* Vh = Vt + (size_t)bh * HEAD_DIM * SEQ;

    int lm  = lane & 15;
    int ka  = (lane < 16) ? 0 : 16;   // A-frag K base
    int kb2 = (lane < 16) ? 0 : 16;   // B-frag K base
    int mb  = (lane < 16) ? 0 : 8;    // C-tile row base

    // Q fragment (A-type, M=query rows, K=head dim 32) - register resident
    FragBF qa;
    {
        const __bf16* Qrow = Qh + (size_t)(m0 + lm) * HEAD_DIM;
#pragma unroll
        for (int v2 = 0; v2 < 8; ++v2) {
            int kk = 2 * v2 + ka;
            qa.e[2 * v2]     = Qrow[kk];
            qa.e[2 * v2 + 1] = Qrow[kk + 1];
        }
    }

    FragF o0, o1;
    float mrow[8], lrow[8];
#pragma unroll
    for (int r = 0; r < 8; ++r) {
        o0.e[r] = 0.0f; o1.e[r] = 0.0f;
        mrow[r] = -1e30f; lrow[r] = 0.0f;
    }
    const float scale = 0.17677669529663687f; // 1/sqrt(32)

    // V staging coordinates: 32 rows (dims) x 64B, row stride SEQ*2 bytes
    int vr = lane >> 2;          // 0..7
    int vc = (lane & 3) * 16;    // 0,16,32,48 bytes

    for (int s0 = 0; s0 < SEQ; s0 += 32) {
        // ---- stage K tile (contiguous 2KB) and V tile (strided rows) ----
        {
            const char* gk = (const char*)(Kh + (size_t)s0 * HEAD_DIM);
            char* lk = (char*)&ktile[wave][0][0];
#pragma unroll
            for (int i = 0; i < 4; ++i)
                copy16_g2l(lk + lane * 16 + i * 512, gk + lane * 16 + i * 512);

            const char* gv = (const char*)(Vh + s0);
            char* lv = (char*)&vtile[wave][0][0];
#pragma unroll
            for (int i = 0; i < 4; ++i) {
                int r = i * 8 + vr;
                copy16_g2l(lv + r * 64 + vc, gv + (size_t)r * (SEQ * 2) + vc);
            }
        }
        async_fence_lds();
        __syncthreads();

        // K^T B-fragments from LDS: B[d][s] = K[s][d]; pairs over consecutive d
        FragBF k0f, k1f;
#pragma unroll
        for (int v2 = 0; v2 < 8; ++v2) {
            int dd = kb2 + 2 * v2;
            k0f.e[2 * v2]     = ktile[wave][lm][dd];
            k0f.e[2 * v2 + 1] = ktile[wave][lm][dd + 1];
            k1f.e[2 * v2]     = ktile[wave][16 + lm][dd];
            k1f.e[2 * v2 + 1] = ktile[wave][16 + lm][dd + 1];
        }

        FragF zf;
#pragma unroll
        for (int r = 0; r < 8; ++r) zf.e[r] = 0.0f;

        FragF s0t, s1t;
        s0t.v = __builtin_amdgcn_wmma_f32_16x16x32_bf16(
            false, qa.v, false, k0f.v, (short)0, zf.v, false, false);
        s1t.v = __builtin_amdgcn_wmma_f32_16x16x32_bf16(
            false, qa.v, false, k1f.v, (short)0, zf.v, false, false);

        // online softmax over this 32-key slab
        float rmax[8];
#pragma unroll
        for (int r = 0; r < 8; ++r) {
            s0t.e[r] *= scale; s1t.e[r] *= scale;
            rmax[r] = fmaxf(s0t.e[r], s1t.e[r]);
        }
#pragma unroll
        for (int off = 1; off < 16; off <<= 1)
#pragma unroll
            for (int r = 0; r < 8; ++r)
                rmax[r] = fmaxf(rmax[r], __shfl_xor(rmax[r], off, 16));

        float rsum[8];
#pragma unroll
        for (int r = 0; r < 8; ++r) {
            float mnew  = fmaxf(mrow[r], rmax[r]);
            float alpha = __expf(mrow[r] - mnew);
            mrow[r] = mnew;
            float p0 = __expf(s0t.e[r] - mnew);
            float p1 = __expf(s1t.e[r] - mnew);
            s0t.e[r] = p0; s1t.e[r] = p1;
            rsum[r] = p0 + p1;
            o0.e[r] *= alpha; o1.e[r] *= alpha;
            lrow[r] *= alpha;
        }
#pragma unroll
        for (int off = 1; off < 16; off <<= 1)
#pragma unroll
            for (int r = 0; r < 8; ++r)
                rsum[r] += __shfl_xor(rsum[r], off, 16);
#pragma unroll
        for (int r = 0; r < 8; ++r) lrow[r] += rsum[r];

        // C-layout f32 P -> LDS -> A-layout bf16 P
#pragma unroll
        for (int r = 0; r < 8; ++r) {
            pbuf[wave][mb + r][lm]      = s0t.e[r];
            pbuf[wave][mb + r][16 + lm] = s1t.e[r];
        }
        __syncthreads();

        FragBF pa;
#pragma unroll
        for (int v2 = 0; v2 < 8; ++v2) {
            int kk = 2 * v2 + ka;
            pa.e[2 * v2]     = (__bf16)pbuf[wave][lm][kk];
            pa.e[2 * v2 + 1] = (__bf16)pbuf[wave][lm][kk + 1];
        }

        // V B-fragments from LDS (vtile is [dim][key]): pairs over consecutive s
        FragBF v0f, v1f;
#pragma unroll
        for (int v2 = 0; v2 < 8; ++v2) {
            int ss = kb2 + 2 * v2;
            v0f.e[2 * v2]     = vtile[wave][lm][ss];
            v0f.e[2 * v2 + 1] = vtile[wave][lm][ss + 1];
            v1f.e[2 * v2]     = vtile[wave][16 + lm][ss];
            v1f.e[2 * v2 + 1] = vtile[wave][16 + lm][ss + 1];
        }
        __syncthreads();

        o0.v = __builtin_amdgcn_wmma_f32_16x16x32_bf16(
            false, pa.v, false, v0f.v, (short)0, o0.v, false, false);
        o1.v = __builtin_amdgcn_wmma_f32_16x16x32_bf16(
            false, pa.v, false, v1f.v, (short)0, o1.v, false, false);
    }

    // write ctx [B,T,D] with softmax normalization
#pragma unroll
    for (int r = 0; r < 8; ++r) {
        float inv = 1.0f / lrow[r];
        size_t row = (size_t)(b * SEQ + m0 + mb + r) * D_MODEL + h * HEAD_DIM;
        ctx[row + lm]      = o0.e[r] * inv;
        ctx[row + 16 + lm] = o1.e[r] * inv;
    }
}

// ---------------------------------------------------------------------------
// Host orchestration
// ---------------------------------------------------------------------------
extern "C" void kernel_launch(void* const* d_in, const int* in_sizes, int n_in,
                              void* d_out, int out_size, void* d_ws, size_t ws_size,
                              hipStream_t stream) {
    (void)in_sizes; (void)n_in; (void)out_size; (void)ws_size;

    const int M  = BATCH * SEQ;          // 4096
    const int NK = D_MODEL;              // 512
    const size_t nAct = (size_t)M * D_MODEL;      // 2,097,152
    const size_t nW   = (size_t)D_MODEL * D_MODEL;// 262,144

    const float* q    = (const float*)d_in[0];
    const float* k    = (const float*)d_in[1];
    const float* v    = (const float*)d_in[2];
    const float* fcos = (const float*)d_in[3];
    const float* fsin = (const float*)d_in[4];
    const float* wq_w = (const float*)d_in[5];
    const float* wq_b = (const float*)d_in[6];
    const float* wk_w = (const float*)d_in[7];
    const float* wk_b = (const float*)d_in[8];
    const float* wv_w = (const float*)d_in[9];
    const float* wv_b = (const float*)d_in[10];
    const float* wo_w = (const float*)d_in[11];
    const float* wo_b = (const float*)d_in[12];
    float* out = (float*)d_out;

    // workspace carve-up (all offsets 256B-aligned by construction)
    char* ws = (char*)d_ws;
    size_t off = 0;
    auto take = [&](size_t bytes) { char* p = ws + off; off += (bytes + 255) & ~(size_t)255; return p; };

    __bf16* Qin = (__bf16*)take(nAct * 2);
    __bf16* Kin = (__bf16*)take(nAct * 2);
    __bf16* Vin = (__bf16*)take(nAct * 2);
    __bf16* Wq  = (__bf16*)take(nW * 2);
    __bf16* Wk  = (__bf16*)take(nW * 2);
    __bf16* Wv  = (__bf16*)take(nW * 2);
    __bf16* Wo  = (__bf16*)take(nW * 2);
    float*  qhF = (float*)take(nAct * 4);
    float*  khF = (float*)take(nAct * 4);
    float*  vhF = (float*)take(nAct * 4);
    __bf16* Qhb = (__bf16*)take(nAct * 2);   // [B,H,T,HD]
    __bf16* Khb = (__bf16*)take(nAct * 2);   // [B,H,T,HD]
    __bf16* Vtb = (__bf16*)take(nAct * 2);   // [B,H,HD,T]
    float*  ctxF = qhF;                      // alias: qhF consumed by rope
    __bf16* ctxB = Qin;                      // alias: Qin consumed by gemm 1

    // 1) f32 -> bf16 conversions
    int cb = 256;
    f32_to_bf16_kernel<<<(int)((nAct + cb - 1) / cb), cb, 0, stream>>>(q, Qin, (int)nAct);
    f32_to_bf16_kernel<<<(int)((nAct + cb - 1) / cb), cb, 0, stream>>>(k, Kin, (int)nAct);
    f32_to_bf16_kernel<<<(int)((nAct + cb - 1) / cb), cb, 0, stream>>>(v, Vin, (int)nAct);
    f32_to_bf16_kernel<<<(int)((nW + cb - 1) / cb), cb, 0, stream>>>(wq_w, Wq, (int)nW);
    f32_to_bf16_kernel<<<(int)((nW + cb - 1) / cb), cb, 0, stream>>>(wk_w, Wk, (int)nW);
    f32_to_bf16_kernel<<<(int)((nW + cb - 1) / cb), cb, 0, stream>>>(wv_w, Wv, (int)nW);
    f32_to_bf16_kernel<<<(int)((nW + cb - 1) / cb), cb, 0, stream>>>(wo_w, Wo, (int)nW);

    // 2) QKV projections: [4096,512] @ [512,512]^T + bias
    dim3 ggrid(NK / 256, M / 64), gblk(256);
    gemm_bf16_wmma_kernel<<<ggrid, gblk, 0, stream>>>(Qin, Wq, wq_b, qhF, M, NK, NK);
    gemm_bf16_wmma_kernel<<<ggrid, gblk, 0, stream>>>(Kin, Wk, wk_b, khF, M, NK, NK);
    gemm_bf16_wmma_kernel<<<ggrid, gblk, 0, stream>>>(Vin, Wv, wv_b, vhF, M, NK, NK);

    // 3) RoPE + head split/pack (+ V transpose)
    int nPairs = BATCH * SEQ * HALF;
    rope_split_kernel<<<(nPairs + cb - 1) / cb, cb, 0, stream>>>(
        qhF, khF, vhF, fcos, fsin, Qhb, Khb, Vtb);

    // 4) flash attention -> ctx f32 [B,T,D]
    int nTiles = BATCH * N_HEADS * (SEQ / 16);               // 4096
    flash_attn_kernel<<<nTiles / ATT_WAVES, 32 * ATT_WAVES, 0, stream>>>(
        Qhb, Khb, Vtb, ctxF);

    // 5) ctx -> bf16, 6) output projection -> d_out
    f32_to_bf16_kernel<<<(int)((nAct + cb - 1) / cb), cb, 0, stream>>>(ctxF, ctxB, (int)nAct);
    gemm_bf16_wmma_kernel<<<ggrid, gblk, 0, stream>>>(ctxB, Wo, wo_b, out, M, NK, NK);
}
